// RelevanceAwarePassageFusion_35708358099450
// MI455X (gfx1250) — matrix operand order repaired and checked by reading, not compiled, not run on hardware
//
#include <hip/hip_runtime.h>
#include <stdint.h>

// Problem constants (match reference)
#define Bsz 4
#define Sl  2048
#define Em  1024
#define Hn  16
#define Dh  64
#define Kb  9

typedef __attribute__((ext_vector_type(16))) __bf16        v16bf;
typedef __attribute__((ext_vector_type(8)))  float         v8f;
typedef __attribute__((ext_vector_type(4)))  unsigned int  u32x4;

typedef __attribute__((address_space(3))) const unsigned short* lds_cptr_t;

union AFrag { v16bf v; u32x4 q[2]; unsigned short us[16]; };
union CFrag { v8f   v; float e[8]; };

__device__ __forceinline__ unsigned short f2bf_u(float f) {
  union { float f; unsigned u; } a; a.f = f;
  unsigned r = a.u + 0x7FFFu + ((a.u >> 16) & 1u);   // round-to-nearest-even
  return (unsigned short)(r >> 16);
}

__device__ __forceinline__ v8f wmma_bf16(v16bf a, v16bf b, v8f c) {
  // D(f32 16x16) = A(bf16 16x32) * B(bf16 32x16) + C
  return __builtin_amdgcn_wmma_f32_16x16x32_bf16(false, a, false, b,
                                                 (short)0, c, false, false);
}

// LDS byte offset of a shared-memory pointer (generic -> AS3 -> int)
__device__ __forceinline__ unsigned lds_off(const unsigned short* p) {
  return (unsigned)(uintptr_t)(lds_cptr_t)p;
}

// Async DMA: 16 bytes global -> LDS per lane (GLOBAL_LOAD_ASYNC_TO_LDS_B128,
// GV mode: 64-bit address in VGPR pair, LDS offset in VDST). ASYNCcnt-tracked.
__device__ __forceinline__ void async_copy_b128(unsigned ldsaddr,
                                                const unsigned short* gptr) {
  unsigned long long ga = (unsigned long long)(uintptr_t)gptr;
  asm volatile("global_load_async_to_lds_b128 %0, %1, off"
               :: "v"(ldsaddr), "v"(ga) : "memory");
}
__device__ __forceinline__ void wait_async0() {
  asm volatile("s_wait_asynccnt 0" ::: "memory");
}

// ---------------------------------------------------------------------------
// Kernel 0: bulk f32 -> bf16 conversion (8 elems/thread, b128 in / b128 out)
// ---------------------------------------------------------------------------
__global__ void f32_to_bf16_kernel(const float* __restrict__ src,
                                   unsigned short* __restrict__ dst, int n8) {
  int i = blockIdx.x * blockDim.x + threadIdx.x;
  if (i >= n8) return;
  const float4* s4 = (const float4*)(src + (size_t)i * 8);
  float4 a = s4[0], b = s4[1];
  union { u32x4 v; unsigned short h[8]; } pk;
  pk.h[0] = f2bf_u(a.x); pk.h[1] = f2bf_u(a.y);
  pk.h[2] = f2bf_u(a.z); pk.h[3] = f2bf_u(a.w);
  pk.h[4] = f2bf_u(b.x); pk.h[5] = f2bf_u(b.y);
  pk.h[6] = f2bf_u(b.z); pk.h[7] = f2bf_u(b.w);
  *(u32x4*)(dst + (size_t)i * 8) = pk.v;
}

// ---------------------------------------------------------------------------
// Kernel 1: per-position passage metadata (replaces the (B,S,S) mask)
// ---------------------------------------------------------------------------
__global__ void prep_mask_kernel(const int* __restrict__ bnd,
                                 const float* __restrict__ rel,
                                 int* __restrict__ pp,
                                 float* __restrict__ colval) {
  int idx = blockIdx.x * blockDim.x + threadIdx.x;
  if (idx >= Bsz * Sl) return;
  int b = idx >> 11, s = idx & (Sl - 1);
  int cnt = 0;
#pragma unroll
  for (int j = 0; j < Kb; ++j) cnt += (bnd[b * Kb + j] <= s) ? 1 : 0;
  int pid = cnt - 1;                              // searchsorted(right) - 1
  bool inp = (pid >= 0) && (pid < Kb - 1);
  pp[idx] = inp ? pid : -1;
  colval[idx] = inp ? rel[b * (Kb - 1) + pid] : 1.0f;
}

// ---------------------------------------------------------------------------
// Kernel 2: QKV projection  (M=8192, N=3072, K=1024, NT, bf16 WMMA)
// Block 256 thr = 8 waves (2x4), tile 64x128, each wave 32x32.
// A/B tiles staged via async global->LDS DMA (pre-converted bf16 inputs).
// ---------------------------------------------------------------------------
__global__ __launch_bounds__(256) void qkv_gemm_kernel(
    const unsigned short* __restrict__ hs_bf, const unsigned short* __restrict__ w_bf,
    const float* __restrict__ bias,
    unsigned short* __restrict__ qws, unsigned short* __restrict__ kws,
    unsigned short* __restrict__ vtws) {
  __shared__ alignas(16) unsigned short ldsA[64][40];
  __shared__ alignas(16) unsigned short ldsB[128][48];
  const int tid = threadIdx.x;
  const int lane = tid & 31, wv = tid >> 5;
  const int wr = wv >> 2, wc = wv & 3;
  const int half = lane >> 4, lc = lane & 15;
  const int m0 = blockIdx.y * 64;
  const int n0 = blockIdx.x * 128;

  const int arow = tid >> 2, ac8 = (tid & 3) * 8;       // A: 1 chunk/thread
  const unsigned ldsA_dst = lds_off(&ldsA[arow][ac8]);
  unsigned ldsB_dst[2];
#pragma unroll
  for (int it = 0; it < 2; ++it) {
    int cc = tid + it * 256;
    ldsB_dst[it] = lds_off(&ldsB[cc >> 2][(cc & 3) * 8]);
  }

  CFrag acc[2][2];
#pragma unroll
  for (int i = 0; i < 2; i++)
#pragma unroll
    for (int j = 0; j < 2; j++)
#pragma unroll
      for (int r = 0; r < 8; r++) acc[i][j].e[r] = 0.f;

  for (int k0 = 0; k0 < Em; k0 += 32) {
    async_copy_b128(ldsA_dst, hs_bf + (size_t)(m0 + arow) * Em + k0 + ac8);
#pragma unroll
    for (int it = 0; it < 2; ++it) {
      int cc = tid + it * 256;
      int row = cc >> 2, c8 = (cc & 3) * 8;
      async_copy_b128(ldsB_dst[it], w_bf + (size_t)(n0 + row) * Em + k0 + c8);
    }
    wait_async0();
    __syncthreads();
    AFrag a[2], bb[2];
#pragma unroll
    for (int mi = 0; mi < 2; mi++) {  // A layout: K chunks {half*8, 16+half*8}
      const unsigned short* p = &ldsA[wr * 32 + mi * 16 + lc][half * 8];
      a[mi].q[0] = *(const u32x4*)p;
      a[mi].q[1] = *(const u32x4*)(p + 16);
    }
#pragma unroll
    for (int ni = 0; ni < 2; ni++) {  // B layout: K chunks {half*8, 16+half*8}
      const unsigned short* p = &ldsB[wc * 32 + ni * 16 + lc][half * 8];
      bb[ni].q[0] = *(const u32x4*)p;
      bb[ni].q[1] = *(const u32x4*)(p + 16);
    }
#pragma unroll
    for (int mi = 0; mi < 2; mi++)
#pragma unroll
      for (int ni = 0; ni < 2; ni++)
        acc[mi][ni].v = wmma_bf16(a[mi].v, bb[ni].v, acc[mi][ni].v);
    __syncthreads();
  }

  // Epilogue: bias, split into Q(scaled)/K/V(transposed), store bf16
#pragma unroll
  for (int mi = 0; mi < 2; mi++)
#pragma unroll
    for (int ni = 0; ni < 2; ni++) {
      int n = n0 + wc * 32 + ni * 16 + lc;
      float bvv = bias[n];
#pragma unroll
      for (int r = 0; r < 8; r++) {
        int m = m0 + wr * 32 + mi * 16 + half * 8 + r;
        int b = m >> 11, s = m & (Sl - 1);
        float val = acc[mi][ni].e[r] + bvv;
        if (n < Em) {
          int h = n >> 6, d = n & 63;
          qws[(((size_t)(b * Hn + h) * Sl + s) << 6) + d] = f2bf_u(val * 0.125f);
        } else if (n < 2 * Em) {
          int e2 = n - Em; int h = e2 >> 6, d = e2 & 63;
          kws[(((size_t)(b * Hn + h) * Sl + s) << 6) + d] = f2bf_u(val);
        } else {
          int e2 = n - 2 * Em; int h = e2 >> 6, d = e2 & 63;
          vtws[(((size_t)(b * Hn + h) * Dh + d) << 11) + s] = f2bf_u(val);
        }
      }
    }
}

// ---------------------------------------------------------------------------
// Kernel 3: flash attention with additive relevance mask.
// Grid (S/64, H, B), 128 threads = 4 waves; each wave owns 16 q rows.
// ---------------------------------------------------------------------------
__global__ __launch_bounds__(128) void attn_kernel(
    const unsigned short* __restrict__ qws, const unsigned short* __restrict__ kws,
    const unsigned short* __restrict__ vtws, const int* __restrict__ pp,
    const float* __restrict__ colval, const int* __restrict__ bnd,
    unsigned short* __restrict__ aows) {
  __shared__ alignas(16) unsigned short pshare[4][16][40];  // per-wave P bounce
  const int tid = threadIdx.x, lane = tid & 31, wv = tid >> 5;
  const int half = lane >> 4, lc = lane & 15;
  const int h = blockIdx.y, b = blockIdx.z;
  const int q0w = blockIdx.x * 64 + wv * 16;
  const size_t headQ = ((size_t)(b * Hn + h) * Sl) << 6;  // elems
  const size_t headV = ((size_t)(b * Hn + h) * Dh) << 11; // elems
  const int qend = bnd[b * Kb];

  AFrag aq[2];  // Q fragments (dh 0..31 / 32..63), loaded once
  {
    const unsigned short* qp = qws + headQ + ((size_t)(q0w + lc) << 6);
#pragma unroll
    for (int kk = 0; kk < 2; kk++) {
      aq[kk].q[0] = *(const u32x4*)(qp + kk * 32 + half * 8);
      aq[kk].q[1] = *(const u32x4*)(qp + kk * 32 + 16 + half * 8);
    }
  }
  int pidq[8], isq[8];
#pragma unroll
  for (int r = 0; r < 8; r++) {
    int qg = q0w + half * 8 + r;
    pidq[r] = pp[b * Sl + qg];
    isq[r] = (qg < qend) ? 1 : 0;
  }
  float m8[8], l8[8];
  CFrag o[4];
#pragma unroll
  for (int r = 0; r < 8; r++) { m8[r] = -1e30f; l8[r] = 0.f; }
#pragma unroll
  for (int t = 0; t < 4; t++)
#pragma unroll
    for (int r = 0; r < 8; r++) o[t].e[r] = 0.f;

  for (int j = 0; j < Sl; j += 32) {
    AFrag bk[2][2];  // K fragments: B-operand col = kv, K-dim = dh chunk
#pragma unroll
    for (int ni = 0; ni < 2; ni++) {
      const unsigned short* kp = kws + headQ + ((size_t)(j + ni * 16 + lc) << 6);
#pragma unroll
      for (int kk = 0; kk < 2; kk++) {
        bk[ni][kk].q[0] = *(const u32x4*)(kp + kk * 32 + half * 8);
        bk[ni][kk].q[1] = *(const u32x4*)(kp + kk * 32 + 16 + half * 8);
      }
    }
    CFrag zz, s0, s1;
#pragma unroll
    for (int r = 0; r < 8; r++) zz.e[r] = 0.f;
    s0.v = wmma_bf16(aq[0].v, bk[0][0].v, zz.v);
    s0.v = wmma_bf16(aq[1].v, bk[0][1].v, s0.v);
    s1.v = wmma_bf16(aq[0].v, bk[1][0].v, zz.v);
    s1.v = wmma_bf16(aq[1].v, bk[1][1].v, s1.v);

    int   pidk0 = pp[b * Sl + j + lc];
    int   pidk1 = pp[b * Sl + j + 16 + lc];
    float ck0   = colval[b * Sl + j + lc];
    float ck1   = colval[b * Sl + j + 16 + lc];

    CFrag corrU;
#pragma unroll
    for (int r = 0; r < 8; r++) {
      s0.e[r] += isq[r] ? ck0 : ((pidq[r] >= 0 && pidq[r] == pidk0) ? 1.0f : 0.0f);
      s1.e[r] += isq[r] ? ck1 : ((pidq[r] >= 0 && pidq[r] == pidk1) ? 1.0f : 0.0f);
      float mx = fmaxf(s0.e[r], s1.e[r]);
#pragma unroll
      for (int msk = 1; msk < 16; msk <<= 1) mx = fmaxf(mx, __shfl_xor(mx, msk, 32));
      float mnew = fmaxf(m8[r], mx);
      float corr = __expf(m8[r] - mnew);
      m8[r] = mnew;
      float p0 = __expf(s0.e[r] - mnew);
      float p1 = __expf(s1.e[r] - mnew);
      float ts = p0 + p1;
#pragma unroll
      for (int msk = 1; msk < 16; msk <<= 1) ts += __shfl_xor(ts, msk, 32);
      l8[r] = l8[r] * corr + ts;
      corrU.e[r] = corr;
      pshare[wv][half * 8 + r][lc]      = f2bf_u(p0);   // C-layout -> LDS
      pshare[wv][half * 8 + r][16 + lc] = f2bf_u(p1);
    }
#pragma unroll
    for (int t = 0; t < 4; t++) o[t].v = o[t].v * corrU.v;

    // Per-wave LDS region; DS ops are in-order per wave, one wait suffices.
    asm volatile("s_wait_dscnt 0" ::: "memory");

    AFrag pf;  // P re-read in A layout (16 q x 32 kv)
    {
      const unsigned short* p = &pshare[wv][lc][half * 8];
      pf.q[0] = *(const u32x4*)p;
      pf.q[1] = *(const u32x4*)(p + 16);
    }
#pragma unroll
    for (int t = 0; t < 4; t++) {  // V fragments from transposed V: contiguous
      AFrag bv;
      const unsigned short* vp = vtws + headV + (((size_t)(t * 16 + lc)) << 11) + j;
      bv.q[0] = *(const u32x4*)(vp + half * 8);
      bv.q[1] = *(const u32x4*)(vp + 16 + half * 8);
      o[t].v = wmma_bf16(pf.v, bv.v, o[t].v);
    }
  }

#pragma unroll
  for (int r = 0; r < 8; r++) l8[r] = 1.0f / l8[r];
#pragma unroll
  for (int t = 0; t < 4; t++)
#pragma unroll
    for (int r = 0; r < 8; r++) {
      int sg = q0w + half * 8 + r;
      int e  = h * Dh + t * 16 + lc;
      aows[((size_t)(b * Sl + sg) << 10) + e] = f2bf_u(o[t].e[r] * l8[r]);
    }
}

// ---------------------------------------------------------------------------
// Kernel 4: output projection (M=8192, N=1024, K=1024, NT), writes f32 d_out.
// ---------------------------------------------------------------------------
__global__ __launch_bounds__(256) void out_gemm_kernel(
    const unsigned short* __restrict__ a_bf, const unsigned short* __restrict__ w_bf,
    const float* __restrict__ bias, float* __restrict__ out) {
  __shared__ alignas(16) unsigned short ldsA[64][40];
  __shared__ alignas(16) unsigned short ldsB[128][48];
  const int tid = threadIdx.x;
  const int lane = tid & 31, wv = tid >> 5;
  const int wr = wv >> 2, wc = wv & 3;
  const int half = lane >> 4, lc = lane & 15;
  const int m0 = blockIdx.y * 64;
  const int n0 = blockIdx.x * 128;

  const int arow = tid >> 2, ac8 = (tid & 3) * 8;
  const unsigned ldsA_dst = lds_off(&ldsA[arow][ac8]);
  unsigned ldsB_dst[2];
#pragma unroll
  for (int it = 0; it < 2; ++it) {
    int cc = tid + it * 256;
    ldsB_dst[it] = lds_off(&ldsB[cc >> 2][(cc & 3) * 8]);
  }

  CFrag acc[2][2];
#pragma unroll
  for (int i = 0; i < 2; i++)
#pragma unroll
    for (int j = 0; j < 2; j++)
#pragma unroll
      for (int r = 0; r < 8; r++) acc[i][j].e[r] = 0.f;

  for (int k0 = 0; k0 < Em; k0 += 32) {
    async_copy_b128(ldsA_dst, a_bf + (size_t)(m0 + arow) * Em + k0 + ac8);
#pragma unroll
    for (int it = 0; it < 2; ++it) {
      int cc = tid + it * 256;
      int row = cc >> 2, c8 = (cc & 3) * 8;
      async_copy_b128(ldsB_dst[it], w_bf + (size_t)(n0 + row) * Em + k0 + c8);
    }
    wait_async0();
    __syncthreads();
    AFrag a[2], bb[2];
#pragma unroll
    for (int mi = 0; mi < 2; mi++) {
      const unsigned short* p = &ldsA[wr * 32 + mi * 16 + lc][half * 8];
      a[mi].q[0] = *(const u32x4*)p;
      a[mi].q[1] = *(const u32x4*)(p + 16);
    }
#pragma unroll
    for (int ni = 0; ni < 2; ni++) {
      const unsigned short* p = &ldsB[wc * 32 + ni * 16 + lc][half * 8];
      bb[ni].q[0] = *(const u32x4*)p;
      bb[ni].q[1] = *(const u32x4*)(p + 16);
    }
#pragma unroll
    for (int mi = 0; mi < 2; mi++)
#pragma unroll
      for (int ni = 0; ni < 2; ni++)
        acc[mi][ni].v = wmma_bf16(a[mi].v, bb[ni].v, acc[mi][ni].v);
    __syncthreads();
  }

#pragma unroll
  for (int mi = 0; mi < 2; mi++)
#pragma unroll
    for (int ni = 0; ni < 2; ni++) {
      int n = n0 + wc * 32 + ni * 16 + lc;
      float bvv = bias[n];
#pragma unroll
      for (int r = 0; r < 8; r++) {
        int m = m0 + wr * 32 + mi * 16 + half * 8 + r;
        out[(size_t)m * Em + n] = acc[mi][ni].e[r] + bvv;
      }
    }
}

// ---------------------------------------------------------------------------
extern "C" void kernel_launch(void* const* d_in, const int* in_sizes, int n_in,
                              void* d_out, int out_size, void* d_ws, size_t ws_size,
                              hipStream_t stream) {
  const float* hs    = (const float*)d_in[0];
  const float* rel   = (const float*)d_in[1];
  const int*   bnd   = (const int*)d_in[2];
  const float* w_in  = (const float*)d_in[3];
  const float* b_in  = (const float*)d_in[4];
  const float* w_out = (const float*)d_in[5];
  const float* b_out = (const float*)d_in[6];
  float* out = (float*)d_out;

  char* ws = (char*)d_ws;
  const size_t qk_elems = (size_t)Bsz * Hn * Sl * Dh;     // 8,388,608
  const size_t hs_elems = (size_t)Bsz * Sl * Em;          // 8,388,608
  const size_t wi_elems = (size_t)3 * Em * Em;            // 3,145,728
  const size_t wo_elems = (size_t)Em * Em;                // 1,048,576
  unsigned short* qws   = (unsigned short*)ws;  ws += qk_elems * 2;
  unsigned short* kws   = (unsigned short*)ws;  ws += qk_elems * 2;
  unsigned short* vtws  = (unsigned short*)ws;  ws += qk_elems * 2;
  unsigned short* aows  = (unsigned short*)ws;  ws += hs_elems * 2;
  unsigned short* hsbf  = (unsigned short*)ws;  ws += hs_elems * 2;
  unsigned short* wibf  = (unsigned short*)ws;  ws += wi_elems * 2;
  unsigned short* wobf  = (unsigned short*)ws;  ws += wo_elems * 2;
  int*   pp  = (int*)ws;   ws += (size_t)Bsz * Sl * 4;
  float* cv  = (float*)ws; ws += (size_t)Bsz * Sl * 4;

  // One-time bf16 staging of GEMM operands (streamed once, then reused).
  f32_to_bf16_kernel<<<(int)(hs_elems / 8 / 256), 256, 0, stream>>>(hs, hsbf,
                                                     (int)(hs_elems / 8));
  f32_to_bf16_kernel<<<(int)(wi_elems / 8 / 256), 256, 0, stream>>>(w_in, wibf,
                                                     (int)(wi_elems / 8));
  f32_to_bf16_kernel<<<(int)(wo_elems / 8 / 256), 256, 0, stream>>>(w_out, wobf,
                                                     (int)(wo_elems / 8));
  prep_mask_kernel<<<(Bsz * Sl + 255) / 256, 256, 0, stream>>>(bnd, rel, pp, cv);

  qkv_gemm_kernel<<<dim3((3 * Em) / 128, (Bsz * Sl) / 64), 256, 0, stream>>>(
      hsbf, wibf, b_in, qws, kws, vtws);
  attn_kernel<<<dim3(Sl / 64, Hn, Bsz), 128, 0, stream>>>(
      qws, kws, vtws, pp, cv, bnd, aows);
  out_gemm_kernel<<<dim3(Em / 128, (Bsz * Sl) / 64), 256, 0, stream>>>(
      aows, wobf, b_out, out);
}
